// SingleHeadAttention_25280177504813
// MI455X (gfx1250) — matrix operand order repaired and checked by reading, not compiled, not run on hardware
//
#include <hip/hip_runtime.h>
#include <math.h>

// Problem constants (match the reference)
#define BATCH  4
#define SLEN   4096
#define DMODEL 1024
#define HEAD   64

typedef __bf16       v16bf __attribute__((ext_vector_type(16)));
typedef float        v8f   __attribute__((ext_vector_type(8)));
typedef unsigned int u32x4 __attribute__((ext_vector_type(4)));
typedef unsigned int u32x2 __attribute__((ext_vector_type(2)));

union Frag { v16bf v; u32x4 u[2]; };

static __device__ inline v8f wmma_bf16(const Frag& a, const Frag& b, v8f c) {
  // D = A(16x32 bf16) * B(32x16 bf16) + C(16x16 f32)
  return __builtin_amdgcn_wmma_f32_16x16x32_bf16(
      /*neg_a=*/false, a.v, /*neg_b=*/false, b.v,
      /*c_mod=*/(short)0, c, /*reuse_a=*/false, /*reuse_b=*/false);
}

// CDNA5 async copy: global -> LDS, 16 bytes per lane, tracked by ASYNCcnt.
static __device__ inline void async_copy_b128(unsigned lds_addr, unsigned long long gaddr) {
  asm volatile("global_load_async_to_lds_b128 %0, %1, off"
               :: "v"(lds_addr), "v"(gaddr) : "memory");
}
static __device__ inline void wait_async0() {
  asm volatile("s_wait_asynccnt 0x0" ::: "memory");
}

// -----------------------------------------------------------------------------
// Kernel 0: one-time W -> bf16 conversion, stored transposed Wt[m][d][k].
// -----------------------------------------------------------------------------
__global__ __launch_bounds__(256)
void w_pack_kernel(const float* __restrict__ Wq, const float* __restrict__ Wk,
                   const float* __restrict__ Wv, __bf16* __restrict__ Wt)
{
  const int idx = blockIdx.x * 256 + threadIdx.x;   // 49152 total: (m,d) x k/4
  const int k4  = (idx & 255) * 4;
  const int dm  = idx >> 8;                         // m*HEAD + d in [0,192)
  const int m   = dm >> 6, d = dm & 63;
  const float* Wp = (m == 0) ? Wq : ((m == 1) ? Wk : Wv);
  union { __bf16 h[4]; u32x2 u; } pk;
#pragma unroll
  for (int j = 0; j < 4; j++) pk.h[j] = (__bf16)Wp[(size_t)(k4 + j) * HEAD + d];
  *(u32x2*)(Wt + (size_t)dm * DMODEL + k4) = pk.u;
}

// -----------------------------------------------------------------------------
// Kernel 1: fused QKV projection + RoPE, fp32 -> bf16.
// Block: 384 threads = 12 waves. wave = matrix(3) x col-tile(4). 64 rows/block.
// Q,K stored row-major bf16 [B*S][64]; V stored transposed [B*64][S].
// Q pre-scaled by HEAD^-0.5 * log2(e) so kernel 2 can use exp2.
// -----------------------------------------------------------------------------
__global__ __launch_bounds__(384)
void qkv_rope_kernel(const float* __restrict__ x,
                     const __bf16* __restrict__ Wt,
                     __bf16* __restrict__ Qw,
                     __bf16* __restrict__ Kw,
                     __bf16* __restrict__ Vt)
{
  __shared__ __attribute__((aligned(16))) __bf16 xs[64 * 32];   // x chunk, row major

  const int tid  = threadIdx.x;
  const int wave = tid >> 5;
  const int lane = tid & 31;
  const int g    = lane >> 4;      // 16-lane half
  const int hn   = lane & 15;
  const int mi   = wave >> 2;      // 0=Q 1=K 2=V
  const int ct   = wave & 3;       // 16-col tile of HEAD
  const int row0 = blockIdx.x * 64;

  const v8f vzero = {0.f,0.f,0.f,0.f,0.f,0.f,0.f,0.f};
  v8f acc[4] = {vzero, vzero, vzero, vzero};

  const __bf16* wrow = Wt + (size_t)(mi * HEAD + ct * 16 + hn) * DMODEL;

  for (int kc = 0; kc < DMODEL; kc += 32) {
    // stage x[64][32] as bf16: thread = 8 consecutive floats -> one b128 LDS store
    if (tid < 256) {
      const int r = tid >> 2, c = (tid & 3) * 8;
      const float* src = x + (size_t)(row0 + r) * DMODEL + kc + c;
      union { __bf16 h[8]; u32x4 u; } pk;
#pragma unroll
      for (int j = 0; j < 8; j++) pk.h[j] = (__bf16)src[j];
      *(u32x4*)&xs[r * 32 + c] = pk.u;
    }
    __syncthreads();

    Frag bfr;   // B: lane=n (out col), K segments at g*8 and 16+g*8 (global)
    bfr.u[0] = *(const u32x4*)(wrow + kc + g * 8);
    bfr.u[1] = *(const u32x4*)(wrow + kc + 16 + g * 8);
#pragma unroll
    for (int rt = 0; rt < 4; rt++) {
      Frag afr; // A: lane row = hn, K segments at g*8 and 16+g*8
      const __bf16* ap = &xs[(rt * 16 + hn) * 32];
      afr.u[0] = *(const u32x4*)(ap + g * 8);
      afr.u[1] = *(const u32x4*)(ap + 16 + g * 8);
      acc[rt] = wmma_bf16(afr, bfr, acc[rt]);
    }
    __syncthreads();
  }

  const int b_idx  = row0 / SLEN;
  const int s_tile = row0 - b_idx * SLEN;

  if (mi < 2) {
    // RoPE on C-layout: row m = r + 8*g, col = ct*16 + hn; pair partner = lane^1
    const int   i     = ct * 8 + (hn >> 1);                       // pair index 0..31
    const float theta = __powf(10000.f, -(float)i * (1.f / 32.f));
    __bf16* dst = (mi == 0) ? Qw : Kw;
    // HEAD^-0.5 * log2(e) folded into Q for base-2 softmax in kernel 2
    const float qscale = (mi == 0) ? (0.125f * 1.44269504088896340736f) : 1.0f;
#pragma unroll
    for (int rt = 0; rt < 4; rt++) {
#pragma unroll
      for (int r = 0; r < 8; r++) {
        const int m   = r + 8 * g;
        const int pos = s_tile + rt * 16 + m;
        float sn, cs;
        __sincosf((float)pos * theta, &sn, &cs);
        const float self = acc[rt][r];
        const float part = __shfl_xor(self, 1, 32);
        const float o = (hn & 1) ? (self * cs + part * sn)        // odd col
                                 : (self * cs - part * sn);       // even col
        dst[(size_t)(row0 + rt * 16 + m) * HEAD + ct * 16 + hn] = (__bf16)(o * qscale);
      }
    }
  } else {
    // V transposed: Vt[(b*64 + dim)*S + s]; 8 consecutive s per lane -> b128 store
#pragma unroll
    for (int rt = 0; rt < 4; rt++) {
      union { __bf16 h[8]; u32x4 u; } pk;
#pragma unroll
      for (int r = 0; r < 8; r++) pk.h[r] = (__bf16)acc[rt][r];
      const size_t base = ((size_t)b_idx * HEAD + ct * 16 + hn) * SLEN
                          + s_tile + rt * 16 + 8 * g;
      *(u32x4*)(Vt + base) = pk.u;
    }
  }
}

// -----------------------------------------------------------------------------
// Kernel 2: causal flash attention, 64-key tiles. Block = 128 threads = 4 waves,
// 64 queries. K/V tiles staged via global_load_async_to_lds_b128 (ASYNCcnt).
// 16 WMMAs per wave per key-tile; base-2 online softmax.
// -----------------------------------------------------------------------------
__global__ __launch_bounds__(128)
void flash_attn_kernel(const __bf16* __restrict__ Qw,
                       const __bf16* __restrict__ Kw,
                       const __bf16* __restrict__ Vt,
                       float* __restrict__ out)
{
  __shared__ __attribute__((aligned(16))) __bf16 ks[64 * HEAD];   // [key][dim]  8KB
  __shared__ __attribute__((aligned(16))) __bf16 vs[HEAD * 64];   // [dim][key]  8KB
  __shared__ __attribute__((aligned(16))) __bf16 ps[4][16 * 64];  // per-wave P  8KB

  const int tid  = threadIdx.x;
  const int wave = tid >> 5, lane = tid & 31;
  const int g    = lane >> 4, hn = lane & 15;
  const int b    = blockIdx.y;
  const int qblk  = blockIdx.x * 64;
  const int qbase = qblk + wave * 16;

  // LDS byte offsets (generic shared pointer low 32 bits == LDS address)
  const unsigned ks_lds = (unsigned)(size_t)(void*)&ks[0];
  const unsigned vs_lds = (unsigned)(size_t)(void*)&vs[0];

  // Q A-fragments (head dims 0..31 and 32..63) straight from global
  Frag qa0, qa1;
  {
    const __bf16* qrow = Qw + (size_t)(b * SLEN + qbase + hn) * HEAD;
    qa0.u[0] = *(const u32x4*)(qrow + g * 8);
    qa0.u[1] = *(const u32x4*)(qrow + 16 + g * 8);
    qa1.u[0] = *(const u32x4*)(qrow + 32 + g * 8);
    qa1.u[1] = *(const u32x4*)(qrow + 48 + g * 8);
  }

  const v8f vzero = {0.f,0.f,0.f,0.f,0.f,0.f,0.f,0.f};
  v8f o[4] = {vzero, vzero, vzero, vzero};
  float mrow[8], lrow[8];
#pragma unroll
  for (int r = 0; r < 8; r++) { mrow[r] = -INFINITY; lrow[r] = 0.f; }

  const int kend = qblk + 63;
  for (int k0 = 0; k0 <= kend; k0 += 64) {
    // async-stage K tile [64][64] and V tile [64][64] (transposed source)
    const __bf16* ksrc = Kw + (size_t)(b * SLEN + k0) * HEAD;
#pragma unroll
    for (int e0 = 0; e0 < 4; e0++) {
      const int e = tid + e0 * 128;                 // e in [0,512)
      async_copy_b128(ks_lds + e * 16,
                      (unsigned long long)(size_t)(ksrc + e * 8));
      const int d = e >> 3, cc = e & 7;
      async_copy_b128(vs_lds + e * 16,
                      (unsigned long long)(size_t)(Vt + (size_t)(b * HEAD + d) * SLEN
                                                   + k0 + cc * 8));
    }
    if (k0 + 64 <= kend)  // pull next K tile toward L2 (global_prefetch_b8)
      __builtin_prefetch(Kw + (size_t)(b * SLEN + k0 + 64) * HEAD + tid * 32, 0, 1);
    wait_async0();
    __syncthreads();

    if (k0 <= qbase + 15) {   // causally relevant for this wave
      // scores: 4 key sub-tiles x (hd 0..31, hd 32..63)
      v8f sacc[4] = {vzero, vzero, vzero, vzero};
#pragma unroll
      for (int st = 0; st < 4; st++) {
        Frag kf; // B = K^T: lane = key, head-dim segments at g*8 and 16+g*8
        const __bf16* kr = &ks[(st * 16 + hn) * HEAD];
        kf.u[0] = *(const u32x4*)(kr + g * 8);
        kf.u[1] = *(const u32x4*)(kr + 16 + g * 8);
        sacc[st] = wmma_bf16(qa0, kf, sacc[st]);
        kf.u[0] = *(const u32x4*)(kr + 32 + g * 8);
        kf.u[1] = *(const u32x4*)(kr + 48 + g * 8);
        sacc[st] = wmma_bf16(qa1, kf, sacc[st]);
      }
      // base-2 online softmax; C-layout: row = r + 8*g, key col = k0 + st*16 + hn
#pragma unroll
      for (int r = 0; r < 8; r++) {
        const int q = qbase + r + 8 * g;
        float xv[4];
#pragma unroll
        for (int st = 0; st < 4; st++)
          xv[st] = (k0 + st * 16 + hn <= q) ? sacc[st][r] : -INFINITY;
        float rm = fmaxf(fmaxf(xv[0], xv[1]), fmaxf(xv[2], xv[3]));
        rm = fmaxf(rm, __shfl_xor(rm, 1, 32));
        rm = fmaxf(rm, __shfl_xor(rm, 2, 32));
        rm = fmaxf(rm, __shfl_xor(rm, 4, 32));
        rm = fmaxf(rm, __shfl_xor(rm, 8, 32));
        const float mn    = fmaxf(mrow[r], rm);
        const float alpha = exp2f(mrow[r] - mn);
        mrow[r] = mn;
        float rs = 0.f;
        __bf16* prow = &ps[wave][(r + 8 * g) * 64];
#pragma unroll
        for (int st = 0; st < 4; st++) {
          const float p = exp2f(xv[st] - mn);
          rs += p;
          prow[st * 16 + hn] = (__bf16)p;
        }
        rs += __shfl_xor(rs, 1, 32);
        rs += __shfl_xor(rs, 2, 32);
        rs += __shfl_xor(rs, 4, 32);
        rs += __shfl_xor(rs, 8, 32);
        lrow[r] = lrow[r] * alpha + rs;
        o[0][r] *= alpha; o[1][r] *= alpha; o[2][r] *= alpha; o[3][r] *= alpha;
      }
      // P (16x64) @ V (64x64): P as two A-fragments from per-wave LDS
      Frag pa0, pa1;
      const __bf16* pr = &ps[wave][hn * 64];
      pa0.u[0] = *(const u32x4*)(pr + g * 8);
      pa0.u[1] = *(const u32x4*)(pr + 16 + g * 8);
      pa1.u[0] = *(const u32x4*)(pr + 32 + g * 8);
      pa1.u[1] = *(const u32x4*)(pr + 48 + g * 8);
#pragma unroll
      for (int t = 0; t < 4; t++) {
        Frag vf; // B = V: lane = head dim, key segments at g*8 and 16+g*8
        const __bf16* vr = &vs[(t * 16 + hn) * 64];
        vf.u[0] = *(const u32x4*)(vr + g * 8);
        vf.u[1] = *(const u32x4*)(vr + 16 + g * 8);
        o[t] = wmma_bf16(pa0, vf, o[t]);
        vf.u[0] = *(const u32x4*)(vr + 32 + g * 8);
        vf.u[1] = *(const u32x4*)(vr + 48 + g * 8);
        o[t] = wmma_bf16(pa1, vf, o[t]);
      }
    }
    __syncthreads();
  }

  // epilogue: O / L -> fp32 out
#pragma unroll
  for (int r = 0; r < 8; r++) {
    const float inv = 1.f / lrow[r];
    const size_t rowoff = (size_t)(b * SLEN + qbase + r + 8 * g) * HEAD;
#pragma unroll
    for (int t = 0; t < 4; t++)
      out[rowoff + t * 16 + hn] = o[t][r] * inv;
  }
}

// -----------------------------------------------------------------------------
extern "C" void kernel_launch(void* const* d_in, const int* in_sizes, int n_in,
                              void* d_out, int out_size, void* d_ws, size_t ws_size,
                              hipStream_t stream) {
  const float* x  = (const float*)d_in[0];
  const float* Wq = (const float*)d_in[1];
  const float* Wk = (const float*)d_in[2];
  const float* Wv = (const float*)d_in[3];

  // Workspace (bf16): Q[B*S][64] | K[B*S][64] | Vt[B*64][S] | Wt[3*64][1024]
  __bf16* Qw = (__bf16*)d_ws;
  __bf16* Kw = Qw + (size_t)BATCH * SLEN * HEAD;
  __bf16* Vt = Kw + (size_t)BATCH * SLEN * HEAD;
  __bf16* Wt = Vt + (size_t)BATCH * SLEN * HEAD;

  w_pack_kernel<<<192, 256, 0, stream>>>(Wq, Wk, Wv, Wt);
  qkv_rope_kernel<<<(BATCH * SLEN) / 64, 384, 0, stream>>>(x, Wt, Qw, Kw, Vt);
  flash_attn_kernel<<<dim3(SLEN / 64, BATCH), 128, 0, stream>>>(Qw, Kw, Vt, (float*)d_out);
}
